// CTIBlock_4956392259962
// MI455X (gfx1250) — matrix operand order repaired and checked by reading, not compiled, not run on hardware
//
#include <hip/hip_runtime.h>
#include <hip/hip_bf16.h>
#include <math.h>

// ---------------- problem constants (B=2, H=W=64 fixed by harness) ----------
#define CDIM   384
#define NHEADS 6
#define DHEAD  64
#define NLVLS  3
#define NPTS   4
#define HIDD   96
#define NBATCH 2
#define HQ     64
#define WQ     64
#define NTOK   21504               // 128*128 + 64*64 + 32*32
#define BNTOK  (NBATCH * NTOK)     // 43008 (multiple of 32)
#define NQTOK  (HQ * WQ)           // 4096
#define OFFD   (NHEADS * NLVLS * NPTS * 2)  // 144
#define AWD    (NHEADS * NLVLS * NPTS)      // 72
#define AWPAD  80                  // 72 padded to tile multiple

// ---------------- CDNA5 WMMA types ------------------------------------------
typedef __attribute__((ext_vector_type(8)))  __bf16 v8bf;
typedef __attribute__((ext_vector_type(16))) __bf16 v16bf;
typedef __attribute__((ext_vector_type(8)))  float  v8f;

// ---------------- helpers ----------------------------------------------------
__device__ __forceinline__ unsigned short f2bf(float f) {
  union { float f; unsigned u; } v; v.f = f;
  unsigned u = v.u + 0x7FFFu + ((v.u >> 16) & 1u);   // round-to-nearest-even
  return (unsigned short)(u >> 16);
}

__device__ __forceinline__ float wave_sum(float x) {
  #pragma unroll
  for (int off = 16; off > 0; off >>= 1) x += __shfl_xor(x, off, 32);
  return x;
}

__device__ __forceinline__ void level_of(int i, int& l, int& start, int& Hl, int& Wl) {
  if (i < 16384)      { l = 0; start = 0;     Hl = 128; Wl = 128; }
  else if (i < 20480) { l = 1; start = 16384; Hl = 64;  Wl = 64;  }
  else                { l = 2; start = 20480; Hl = 32;  Wl = 32;  }
}

__device__ __forceinline__ v16bf load_afrag(const unsigned short* p) {
  v8bf lo = *(const v8bf*)(const void*)(p);
  v8bf hi = *(const v8bf*)(const void*)(p + 8);
  return __builtin_shufflevector(lo, hi, 0,1,2,3,4,5,6,7,8,9,10,11,12,13,14,15);
}

// ---------------- weight cast + transpose to bf16 [Npad rows x K] ------------
__global__ void wcast_kernel(const float* __restrict__ W, unsigned short* __restrict__ Wt,
                             int K, int N, int Npad) {
  int id = blockIdx.x * blockDim.x + threadIdx.x;
  if (id >= Npad * K) return;
  int n = id / K, k = id % K;
  float v = (n < N) ? W[(size_t)k * N + n] : 0.f;
  Wt[id] = f2bf(v);
}

// ---------------- LayerNorm: wave32 per token, dual-output optional ----------
__global__ void layernorm_kernel(const float* __restrict__ x,
                                 const float* __restrict__ g0, const float* __restrict__ bb0,
                                 const float* __restrict__ g1, const float* __restrict__ bb1,
                                 unsigned short* __restrict__ o0,
                                 unsigned short* __restrict__ o1, int T) {
  int wid  = (blockIdx.x * blockDim.x + threadIdx.x) >> 5;
  int lane = threadIdx.x & 31;
  if (wid >= T) return;
  const float* xp = x + (size_t)wid * CDIM;
  float xs[12]; float s = 0.f;
  #pragma unroll
  for (int j = 0; j < 12; ++j) { xs[j] = xp[lane + j * 32]; s += xs[j]; }
  s = wave_sum(s);
  float mu = s * (1.0f / CDIM);
  float var = 0.f;
  #pragma unroll
  for (int j = 0; j < 12; ++j) { float d = xs[j] - mu; var += d * d; }
  var = wave_sum(var) * (1.0f / CDIM);
  float rs = rsqrtf(var + 1e-6f);
  #pragma unroll
  for (int j = 0; j < 12; ++j) {
    int c = lane + j * 32;
    float xn = (xs[j] - mu) * rs;
    o0[(size_t)wid * CDIM + c] = f2bf(xn * g0[c] + bb0[c]);
    if (o1) o1[(size_t)wid * CDIM + c] = f2bf(xn * g1[c] + bb1[c]);
  }
}

// ---------------- generic bf16 WMMA GEMM: one 16x16 tile per wave ------------
// Used for the narrow-N projections (N = 144 / 80 / 96).
__global__ void gemm_bf16_wmma(const unsigned short* __restrict__ A,
                               const unsigned short* __restrict__ Bt,
                               const float* __restrict__ bias,
                               float* __restrict__ Cmat,
                               int M, int N, int Npad, int K, int accumulate) {
  const int lane = threadIdx.x & 31;
  const int wave = threadIdx.x >> 5;
  const int Nt = Npad >> 4;
  const int Mt = M >> 4;
  int tile = blockIdx.x * 8 + wave;            // wave-uniform
  if (tile >= Mt * Nt) return;                 // wave-uniform exit: EXEC stays all-ones
  const int mt = tile / Nt, nt = tile % Nt;
  const int row0 = mt << 4, col0 = nt << 4;
  const int lh = lane & 15, kh = lane >> 4;

  const unsigned short* arow = A  + (size_t)(row0 + lh) * K + kh * 16;
  const unsigned short* bcol = Bt + (size_t)(col0 + lh) * K + kh * 16;

  v8f acc = {};
  for (int k0 = 0; k0 < K; k0 += 32) {
    v16bf a = load_afrag(arow + k0);
    v16bf b = *(const v16bf*)(const void*)(bcol + k0);
    acc = __builtin_amdgcn_wmma_f32_16x16x32_bf16(false, a, false, b,
                                                  (short)0, acc, false, false);
  }
  const int col = col0 + lh;
  if (col < N) {
    float bvv = bias ? bias[col] : 0.f;
    #pragma unroll
    for (int r = 0; r < 8; ++r) {
      int row = row0 + r + kh * 8;             // C/D layout: VGPR r -> M = r (+8 hi half)
      size_t o = (size_t)row * N + col;
      float v = acc[r] + bvv;
      if (accumulate) v += Cmat[o];
      Cmat[o] = v;
    }
  }
}

// ---------------- register-blocked WMMA GEMM: 32x64 tile per wave ------------
// Requires M % 32 == 0 and N % 64 == 0 (the three N=384 GEMMs).
// 2 M-tiles x 4 N-tiles: A traffic /4, B traffic /2, 8 WMMAs per K-step.
__global__ void gemm_bf16_wmma_2x4(const unsigned short* __restrict__ A,
                                   const unsigned short* __restrict__ Bt,
                                   const float* __restrict__ bias,
                                   float* __restrict__ Cmat,
                                   int M, int N, int K, int accumulate) {
  const int lane = threadIdx.x & 31;
  const int wave = threadIdx.x >> 5;
  const int Nt = N >> 6;
  const int Mt = M >> 5;
  int tile = blockIdx.x * 8 + wave;            // wave-uniform
  if (tile >= Mt * Nt) return;                 // EXEC stays all-ones at the WMMAs
  const int mt = tile / Nt, nt = tile % Nt;
  const int row0 = mt << 5, col0 = nt << 6;
  const int lh = lane & 15, kh = lane >> 4;

  const unsigned short* arow0 = A + (size_t)(row0 + lh) * K + kh * 16;
  const unsigned short* arow1 = arow0 + (size_t)16 * K;
  const unsigned short* bcol  = Bt + (size_t)(col0 + lh) * K + kh * 16;

  v8f acc[2][4] = {};
  for (int k0 = 0; k0 < K; k0 += 32) {
    v16bf a0 = load_afrag(arow0 + k0);
    v16bf a1 = load_afrag(arow1 + k0);
    #pragma unroll
    for (int j = 0; j < 4; ++j) {
      v16bf b = *(const v16bf*)(const void*)(bcol + (size_t)j * 16 * K + k0);
      acc[0][j] = __builtin_amdgcn_wmma_f32_16x16x32_bf16(false, a0, false, b,
                                                          (short)0, acc[0][j], false, false);
      acc[1][j] = __builtin_amdgcn_wmma_f32_16x16x32_bf16(false, a1, false, b,
                                                          (short)0, acc[1][j], false, false);
    }
  }
  #pragma unroll
  for (int j = 0; j < 4; ++j) {
    int col = col0 + j * 16 + lh;
    float bvv = bias ? bias[col] : 0.f;
    #pragma unroll
    for (int i = 0; i < 2; ++i) {
      #pragma unroll
      for (int r = 0; r < 8; ++r) {
        int row = row0 + i * 16 + r + kh * 8;
        size_t o = (size_t)row * N + col;
        float v = acc[i][j][r] + bvv;
        if (accumulate) v += Cmat[o];
        Cmat[o] = v;
      }
    }
  }
}

// ---------------- softmax over 12 points per (token, head) -------------------
__global__ void softmax12_kernel(float* __restrict__ aw) {
  int t = blockIdx.x * blockDim.x + threadIdx.x;
  if (t >= BNTOK * NHEADS) return;
  float* p = aw + (size_t)(t / NHEADS) * AWD + (size_t)(t % NHEADS) * 12;
  float m = p[0];
  #pragma unroll
  for (int j = 1; j < 12; ++j) m = fmaxf(m, p[j]);
  float e[12]; float s = 0.f;
  #pragma unroll
  for (int j = 0; j < 12; ++j) { e[j] = __expf(p[j] - m); s += e[j]; }
  float inv = 1.f / s;
  #pragma unroll
  for (int j = 0; j < 12; ++j) p[j] = e[j] * inv;
}

// ---------------- multi-scale deformable attention ---------------------------
// One wave per (batch, token, head); lane covers channels {lane, lane+32}.
__global__ void deform_attn_kernel(const float* __restrict__ val,
                                   const float* __restrict__ off,
                                   const float* __restrict__ aw,
                                   unsigned short* __restrict__ attn) {
  int wid  = (blockIdx.x * blockDim.x + threadIdx.x) >> 5;
  int lane = threadIdx.x & 31;
  if (wid >= BNTOK * NHEADS) return;
  int h = wid % NHEADS;
  int t = wid / NHEADS;            // b*NTOK + i
  int b = t / NTOK, i = t % NTOK;
  int ql, qs, qH, qW; level_of(i, ql, qs, qH, qW);
  int p = i - qs; int py = p / qW, px = p % qW;
  float refx = (px + 0.5f) / (float)qW;
  float refy = (py + 0.5f) / (float)qH;

  const float* offp = off + (size_t)t * OFFD + (size_t)h * (NLVLS * NPTS * 2);
  const float* awp  = aw  + (size_t)t * AWD  + (size_t)h * (NLVLS * NPTS);

  float o0 = 0.f, o1 = 0.f;
  for (int l = 0; l < NLVLS; ++l) {
    int Hl = (l == 0) ? 128 : (l == 1 ? 64 : 32);
    int Wl = Hl;
    int start = (l == 0) ? 0 : (l == 1 ? 16384 : 20480);
    const float* vbase = val + ((size_t)b * NTOK + start) * CDIM + (size_t)h * DHEAD;
    for (int pt = 0; pt < NPTS; ++pt) {
      float w  = awp[l * NPTS + pt];
      float xf = (refx + offp[(l * NPTS + pt) * 2 + 0] / (float)Wl) * Wl - 0.5f;
      float yf = (refy + offp[(l * NPTS + pt) * 2 + 1] / (float)Hl) * Hl - 0.5f;
      float x0 = floorf(xf), y0 = floorf(yf);
      #pragma unroll
      for (int dy = 0; dy < 2; ++dy) {
        #pragma unroll
        for (int dx = 0; dx < 2; ++dx) {
          float xi = x0 + dx, yi = y0 + dy;
          if (xi < 0.f || xi >= (float)Wl || yi < 0.f || yi >= (float)Hl) continue;
          float wc = w * (1.f - fabsf(xf - xi)) * (1.f - fabsf(yf - yi));
          const float* vp = vbase + (size_t)((int)yi * Wl + (int)xi) * CDIM;
          o0 += wc * vp[lane];
          o1 += wc * vp[lane + 32];
        }
      }
    }
  }
  unsigned short* op = attn + (size_t)t * CDIM + (size_t)h * DHEAD;
  op[lane]      = f2bf(o0);
  op[lane + 32] = f2bf(o1);
}

// ---------------- depthwise 3x3 conv (SAME, per scale) + bias + exact GELU ---
__global__ void dwconv_gelu_kernel(const float* __restrict__ h,
                                   const float* __restrict__ dw_w,
                                   const float* __restrict__ dw_b,
                                   unsigned short* __restrict__ hg) {
  int id = blockIdx.x * blockDim.x + threadIdx.x;
  if (id >= BNTOK * HIDD) return;
  int ch = id % HIDD;
  int t  = id / HIDD;
  int b = t / NTOK, i = t % NTOK;
  int l, start, Hl, Wl; level_of(i, l, start, Hl, Wl);
  int p = i - start; int y = p / Wl, x = p % Wl;
  float acc = 0.f;
  #pragma unroll
  for (int dy = -1; dy <= 1; ++dy) {
    int yy = y + dy;
    if (yy < 0 || yy >= Hl) continue;
    #pragma unroll
    for (int dx = -1; dx <= 1; ++dx) {
      int xx = x + dx;
      if (xx < 0 || xx >= Wl) continue;
      float w = dw_w[((dy + 1) * 3 + (dx + 1)) * HIDD + ch];
      acc += h[(size_t)(b * NTOK + start + yy * Wl + xx) * HIDD + ch] * w;
    }
  }
  acc += dw_b[ch];
  float g = 0.5f * acc * (1.f + erff(acc * 0.70710678118654752f));
  hg[(size_t)t * HIDD + ch] = f2bf(g);
}

// ---------------- multi-scale fuse: query + gamma*(avgpool + id + bilinear) --
__global__ void combine_kernel(const float* __restrict__ query,
                               const float* __restrict__ gamma,
                               const float* __restrict__ c1,
                               float* __restrict__ out) {
  int id = blockIdx.x * blockDim.x + threadIdx.x;
  if (id >= NBATCH * NQTOK * CDIM) return;
  int c = id % CDIM;
  int r = id / CDIM;
  int x = r % WQ;
  int y = (r / WQ) % HQ;
  int b = r / (WQ * HQ);
  const float* base = c1 + (size_t)b * NTOK * CDIM;
  // s1: 2x2 mean over the 128x128 region
  float s1 = 0.f;
  #pragma unroll
  for (int dy = 0; dy < 2; ++dy)
    #pragma unroll
    for (int dx = 0; dx < 2; ++dx)
      s1 += base[(size_t)((2 * y + dy) * 128 + (2 * x + dx)) * CDIM + c];
  s1 *= 0.25f;
  // s2: identity from 64x64 region
  float s2 = base[(size_t)(16384 + y * 64 + x) * CDIM + c];
  // s3: bilinear 32x32 -> 64x64 (half-pixel centers, edge clamp)
  float sy = y * 0.5f - 0.25f;
  float sx = x * 0.5f - 0.25f;
  float fy = floorf(sy), fx = floorf(sx);
  float wy = sy - fy,   wx = sx - fx;
  int y0 = max(0, min(31, (int)fy)), y1 = max(0, min(31, (int)fy + 1));
  int x0 = max(0, min(31, (int)fx)), x1 = max(0, min(31, (int)fx + 1));
  const float* s3b = base + (size_t)20480 * CDIM;
  float v00 = s3b[(size_t)(y0 * 32 + x0) * CDIM + c];
  float v01 = s3b[(size_t)(y0 * 32 + x1) * CDIM + c];
  float v10 = s3b[(size_t)(y1 * 32 + x0) * CDIM + c];
  float v11 = s3b[(size_t)(y1 * 32 + x1) * CDIM + c];
  float s3 = (1.f - wy) * ((1.f - wx) * v00 + wx * v01)
           +        wy  * ((1.f - wx) * v10 + wx * v11);
  out[id] = query[id] + gamma[c] * (s1 + s2 + s3);
}

// ---------------- host orchestration ----------------------------------------
extern "C" void kernel_launch(void* const* d_in, const int* in_sizes, int n_in,
                              void* d_out, int out_size, void* d_ws, size_t ws_size,
                              hipStream_t stream) {
  (void)in_sizes; (void)n_in; (void)out_size; (void)ws_size;
  const float* query = (const float*)d_in[0];
  const float* feat  = (const float*)d_in[1];
  const float* qn_g  = (const float*)d_in[2];
  const float* qn_b  = (const float*)d_in[3];
  const float* fn_g  = (const float*)d_in[4];
  const float* fn_b  = (const float*)d_in[5];
  const float* Wv    = (const float*)d_in[6];
  const float* bv    = (const float*)d_in[7];
  const float* Woff  = (const float*)d_in[8];
  const float* boff  = (const float*)d_in[9];
  const float* Wattn = (const float*)d_in[10];
  const float* battn = (const float*)d_in[11];
  const float* Wout  = (const float*)d_in[12];
  const float* bout  = (const float*)d_in[13];
  const float* gamma = (const float*)d_in[14];
  const float* ffn_g = (const float*)d_in[15];
  const float* ffn_b = (const float*)d_in[16];
  const float* W1    = (const float*)d_in[17];
  const float* b1    = (const float*)d_in[18];
  const float* dw_w  = (const float*)d_in[19];
  const float* dw_b  = (const float*)d_in[20];
  const float* W2    = (const float*)d_in[21];
  const float* b2    = (const float*)d_in[22];

  char* ws = (char*)d_ws;
  size_t cur = 0;
  auto alloc = [&](size_t bytes) -> char* {
    char* p = ws + cur;
    cur = (cur + bytes + 255) & ~(size_t)255;
    return p;
  };
  unsigned short* q_bf    = (unsigned short*)alloc((size_t)BNTOK * CDIM * 2);
  unsigned short* vln_bf  = (unsigned short*)alloc((size_t)BNTOK * CDIM * 2);
  float*          v_f     = (float*)alloc((size_t)BNTOK * CDIM * 4);
  float*          off_f   = (float*)alloc((size_t)BNTOK * OFFD * 4);
  float*          aw_f    = (float*)alloc((size_t)BNTOK * AWD * 4);
  unsigned short* attn_bf = (unsigned short*)alloc((size_t)BNTOK * CDIM * 2);
  float*          c1_f    = (float*)alloc((size_t)BNTOK * CDIM * 4);
  unsigned short* ln2_bf  = (unsigned short*)alloc((size_t)BNTOK * CDIM * 2);
  float*          h_f     = (float*)alloc((size_t)BNTOK * HIDD * 4);
  unsigned short* hg_bf   = (unsigned short*)alloc((size_t)BNTOK * HIDD * 2);
  unsigned short* Wv_t    = (unsigned short*)alloc((size_t)CDIM * CDIM * 2);
  unsigned short* Woff_t  = (unsigned short*)alloc((size_t)OFFD * CDIM * 2);
  unsigned short* Wattn_t = (unsigned short*)alloc((size_t)AWPAD * CDIM * 2);
  unsigned short* Wout_t  = (unsigned short*)alloc((size_t)CDIM * CDIM * 2);
  unsigned short* W1_t    = (unsigned short*)alloc((size_t)HIDD * CDIM * 2);
  unsigned short* W2_t    = (unsigned short*)alloc((size_t)CDIM * HIDD * 2);

  // 1) bf16 weight casts (transposed to [N][K] so B-frags are contiguous)
  wcast_kernel<<<(CDIM * CDIM + 255) / 256, 256, 0, stream>>>(Wv,   Wv_t,   CDIM, CDIM, CDIM);
  wcast_kernel<<<(OFFD * CDIM + 255) / 256, 256, 0, stream>>>(Woff, Woff_t, CDIM, OFFD, OFFD);
  wcast_kernel<<<(AWPAD * CDIM + 255) / 256, 256, 0, stream>>>(Wattn, Wattn_t, CDIM, AWD, AWPAD);
  wcast_kernel<<<(CDIM * CDIM + 255) / 256, 256, 0, stream>>>(Wout, Wout_t, CDIM, CDIM, CDIM);
  wcast_kernel<<<(HIDD * CDIM + 255) / 256, 256, 0, stream>>>(W1,   W1_t,   CDIM, HIDD, HIDD);
  wcast_kernel<<<(CDIM * HIDD + 255) / 256, 256, 0, stream>>>(W2,   W2_t,   HIDD, CDIM, CDIM);

  // 2) dual LayerNorm on feat -> q (bf16), v-input (bf16)
  layernorm_kernel<<<BNTOK / 8, 256, 0, stream>>>(feat, qn_g, qn_b, fn_g, fn_b,
                                                  q_bf, vln_bf, BNTOK);

  auto gblocks1 = [](int M, int Npad) { return ((M / 16) * (Npad / 16) + 7) / 8; };
  auto gblocks2 = [](int M, int N)    { return ((M / 32) * (N / 64) + 7) / 8; };

  // 3) projections (value uses the register-blocked kernel, N=384)
  gemm_bf16_wmma_2x4<<<gblocks2(BNTOK, CDIM), 256, 0, stream>>>(vln_bf, Wv_t, bv, v_f,
                                                                BNTOK, CDIM, CDIM, 0);
  gemm_bf16_wmma<<<gblocks1(BNTOK, OFFD), 256, 0, stream>>>(q_bf, Woff_t, boff, off_f,
                                                            BNTOK, OFFD, OFFD, CDIM, 0);
  gemm_bf16_wmma<<<gblocks1(BNTOK, AWPAD), 256, 0, stream>>>(q_bf, Wattn_t, battn, aw_f,
                                                             BNTOK, AWD, AWPAD, CDIM, 0);
  // 4) softmax + deformable attention
  softmax12_kernel<<<(BNTOK * NHEADS + 255) / 256, 256, 0, stream>>>(aw_f);
  deform_attn_kernel<<<BNTOK * NHEADS / 8, 256, 0, stream>>>(v_f, off_f, aw_f, attn_bf);
  // 5) output projection -> c1 (N=384, register-blocked)
  gemm_bf16_wmma_2x4<<<gblocks2(BNTOK, CDIM), 256, 0, stream>>>(attn_bf, Wout_t, bout, c1_f,
                                                                BNTOK, CDIM, CDIM, 0);
  // 6) conv-FFN: LN -> W1 -> dwconv+GELU -> W2 (accumulated into c1)
  layernorm_kernel<<<BNTOK / 8, 256, 0, stream>>>(c1_f, ffn_g, ffn_b, nullptr, nullptr,
                                                  ln2_bf, nullptr, BNTOK);
  gemm_bf16_wmma<<<gblocks1(BNTOK, HIDD), 256, 0, stream>>>(ln2_bf, W1_t, b1, h_f,
                                                            BNTOK, HIDD, HIDD, CDIM, 0);
  dwconv_gelu_kernel<<<(BNTOK * HIDD + 255) / 256, 256, 0, stream>>>(h_f, dw_w, dw_b, hg_bf);
  gemm_bf16_wmma_2x4<<<gblocks2(BNTOK, CDIM), 256, 0, stream>>>(hg_bf, W2_t, b2, c1_f,
                                                                BNTOK, CDIM, HIDD, 1);
  // 7) multi-scale fuse + residual
  combine_kernel<<<(NBATCH * NQTOK * CDIM + 255) / 256, 256, 0, stream>>>(
      query, gamma, c1_f, (float*)d_out);
}
